// STFAGCN_72164040507791
// MI455X (gfx1250) — matrix operand order repaired and checked by reading, not compiled, not run on hardware
//
#include <hip/hip_runtime.h>
#include <math.h>

// ---------------------------------------------------------------------------
// CDNA5 (gfx1250) wave32 WMMA types / helpers
// ---------------------------------------------------------------------------
typedef __attribute__((ext_vector_type(16))) __bf16       v16bf;
typedef __attribute__((ext_vector_type(8)))  float        v8f;
typedef __attribute__((ext_vector_type(4)))  unsigned int v4u;

union BFrag {
    v4u            q[2];
    unsigned short u[16];
    v16bf          v;
};

// native fptrunc: lowers to gfx1250 bf16 cvt hardware (RNE), not SW emulation
__device__ __forceinline__ unsigned short f2bf(float f) {
    __bf16 h = (__bf16)f;
    return __builtin_bit_cast(unsigned short, h);
}
__device__ __forceinline__ float bf2f(unsigned short h) {
    return __uint_as_float(((unsigned int)h) << 16);
}

// ---------------------------------------------------------------------------
// Feature extractor. ONE persistent wave32 per block, grid-stride over nodes.
// All three conv/pool stages run on the WMMA pipe; weights live in VGPRs for
// the wave lifetime; biases ride in the WMMA C operand.
//   conv1: D1[32oc][64px] = W1pad[32x32] @ xpatch[32x64]      (8 WMMA)
//          relu(D1) scattered into conv2 im2col patch[pixel][K]
//   conv2: D2[64oc][96px] = W2[64x128] @ patch[128x96]        (96 WMMA)
//   pool : pooled[64x4]   = relu(D2)[64x96] @ P[96x4]         (12 WMMA)
//   FC 256 -> 32 (VALU, lane = output feature)
// ---------------------------------------------------------------------------
#define PATCH_LD 136   // 128 + 8 halves: 272B row stride -> conflict-free b128
#define C2_LD    104   // 96 + 8 halves
#define XP_LD    40    // 32 + 8 halves

__global__ __launch_bounds__(32) void fe_kernel(
    const float* __restrict__ x, const float* __restrict__ c1w,
    const float* __restrict__ c1b, const float* __restrict__ c2w,
    const float* __restrict__ c2b, const float* __restrict__ few,
    const float* __restrict__ feb, unsigned short* __restrict__ sigbf, int n)
{
    __shared__ unsigned short xpatch[64 * XP_LD];    // conv1 im2col [pixel][K<=32]
    __shared__ unsigned short patch[96 * PATCH_LD];  // conv2 im2col [pixel][K] bf16
    __shared__ unsigned short c2lds[64 * C2_LD];     // relu(conv2) [oc][pixel] bf16
    __shared__ float          pooled[256];           // [oc][bin]
    __shared__ __attribute__((aligned(16))) float xs[64];

    const int lane  = threadIdx.x;
    const int pcol  = lane & 15;
    const int lhalf = lane >> 4;

    // ---- wave-lifetime register state ------------------------------------
    // conv1 weight A-fragments (K padded 9 -> 32 with zeros)
    BFrag wfr1[2];
    #pragma unroll
    for (int mt = 0; mt < 2; ++mt) {
        const int oc = mt * 16 + pcol;
        #pragma unroll
        for (int j = 0; j < 16; ++j) {
            const int K = lhalf * 16 + j;
            wfr1[mt].u[j] = (K < 9) ? f2bf(c1w[oc * 9 + K]) : (unsigned short)0;
        }
    }
    v8f cbias1[2];
    #pragma unroll
    for (int mt = 0; mt < 2; ++mt)
        #pragma unroll
        for (int v = 0; v < 8; ++v)
            cbias1[mt][v] = c1b[mt * 16 + lhalf * 8 + v];

    // conv2 weight A-fragments: wfr[mt][ks], row oc = mt*16+pcol,
    // K(j) = ks*32 + lhalf*16 + j
    BFrag wfr[4][4];
    #pragma unroll
    for (int mt = 0; mt < 4; ++mt)
        #pragma unroll
        for (int ks = 0; ks < 4; ++ks) {
            const int oc = mt * 16 + pcol;
            #pragma unroll
            for (int j = 0; j < 16; ++j) {
                const int K = ks * 32 + lhalf * 16 + j;
                wfr[mt][ks].u[j] = f2bf(c2w[oc * 128 + K]);
            }
        }
    v8f cbias[4];
    #pragma unroll
    for (int mt = 0; mt < 4; ++mt)
        #pragma unroll
        for (int v = 0; v < 8; ++v)
            cbias[mt][v] = c2b[mt * 16 + lhalf * 8 + v];

    // pooling matrix B-fragments (P[96x4], weights 0 or 1/32)
    BFrag pfr[3];
    #pragma unroll
    for (int ks = 0; ks < 3; ++ks) {
        const int kb = ks * 32 + lhalf * 16;
        #pragma unroll
        for (int j = 0; j < 16; ++j) {
            const int p = kb + j;
            const int py = p / 9, px = p - py * 9;
            float wgt = 0.0f;
            if (pcol < 4 && p < 81) {
                const bool oy = (pcol & 2) ? (py >= 4) : (py <= 4);
                const bool ox = (pcol & 1) ? (px >= 4) : (px <= 4);
                wgt = (oy && ox) ? 0.03125f : 0.0f;      // 1/32, exact in bf16
            }
            pfr[ks].u[j] = f2bf(wgt);
        }
    }

    // zero im2col buffers ONCE: interiors are rewritten every node, halos and
    // padded-K columns stay zero forever
    {
        v4u z = {0u, 0u, 0u, 0u};
        v4u* pz = (v4u*)patch;
        for (int i = lane; i < (96 * PATCH_LD) / 8; i += 32) pz[i] = z;
        v4u* xz = (v4u*)xpatch;
        for (int i = lane; i < (64 * XP_LD) / 8; i += 32) xz[i] = z;
    }
    __builtin_amdgcn_wave_barrier();

    for (int node = blockIdx.x; node < n; node += gridDim.x) {
        // ---- async-LDS load of the node's 8x8 input tile (ASYNCcnt path) ----
        {
            unsigned           ldsa = (unsigned)(unsigned long long)(const void*)&xs[2 * lane];
            unsigned long long ga   = (unsigned long long)(const void*)&x[(size_t)node * 64 + 2 * lane];
            asm volatile(
                "global_load_async_to_lds_b64 %0, %1, off\n\t"
                "s_wait_asynccnt 0"
                :: "v"(ldsa), "v"(ga) : "memory");
        }
        __builtin_amdgcn_wave_barrier();

        // ---- scatter input tile into conv1 im2col xpatch ----
        #pragma unroll
        for (int pp = 0; pp < 2; ++pp) {
            const int p  = lane + pp * 32;
            const int iy = p >> 3, ix = p & 7;
            const unsigned short bf = f2bf(xs[p]);
            #pragma unroll
            for (int ky = 0; ky < 3; ++ky) {
                const int y = iy + 1 - ky;
                if ((unsigned)y < 8u) {
                    #pragma unroll
                    for (int kx = 0; kx < 3; ++kx) {
                        const int xc = ix + 1 - kx;
                        if ((unsigned)xc < 8u)
                            xpatch[(y * 8 + xc) * XP_LD + ky * 3 + kx] = bf;
                    }
                }
            }
        }
        __builtin_amdgcn_wave_barrier();

        // ---- conv1 WMMA; relu(D1) scattered into conv2 im2col patch ----
        for (int nt = 0; nt < 4; ++nt) {
            const int p = nt * 16 + pcol;            // conv1 output pixel 0..63
            BFrag bx;
            const unsigned short* pr = &xpatch[p * XP_LD + lhalf * 16];
            bx.q[0] = *(const v4u*)pr;
            bx.q[1] = *(const v4u*)(pr + 8);
            const int y = p >> 3, xc = p & 7;
            #pragma unroll
            for (int mt = 0; mt < 2; ++mt) {
                v8f acc = cbias1[mt];
                acc = __builtin_amdgcn_wmma_f32_16x16x32_bf16(
                        false, wfr1[mt].v, false, bx.v, (short)0, acc, false, false);
                #pragma unroll
                for (int v = 0; v < 8; ++v) {
                    const int oc = mt * 16 + lhalf * 8 + v;
                    const unsigned short bf = f2bf(fmaxf(acc[v], 0.0f));
                    // input (y,x) feeds out-pixel (y+1-kh, x+1-kw); K=oc*4+kh*2+kw
                    patch[((y + 1) * 9 + (xc + 1)) * PATCH_LD + oc * 4 + 0] = bf;
                    patch[((y + 1) * 9 +  xc     ) * PATCH_LD + oc * 4 + 1] = bf;
                    patch[( y      * 9 + (xc + 1)) * PATCH_LD + oc * 4 + 2] = bf;
                    patch[( y      * 9 +  xc     ) * PATCH_LD + oc * 4 + 3] = bf;
                }
            }
        }
        __builtin_amdgcn_wave_barrier();

        // ---- conv2 implicit GEMM; stage relu result bf16 to LDS ----
        for (int nt = 0; nt < 6; ++nt) {
            const int p = nt * 16 + pcol;            // rows 81..95 are all-zero
            BFrag bfr[4];
            #pragma unroll
            for (int ks = 0; ks < 4; ++ks) {         // 16 contiguous halves
                const unsigned short* pr2 = &patch[p * PATCH_LD + ks * 32 + lhalf * 16];
                bfr[ks].q[0] = *(const v4u*)pr2;
                bfr[ks].q[1] = *(const v4u*)(pr2 + 8);
            }
            #pragma unroll
            for (int mt = 0; mt < 4; ++mt) {
                v8f acc = cbias[mt];                 // bias via C operand
                #pragma unroll
                for (int ks = 0; ks < 4; ++ks)
                    acc = __builtin_amdgcn_wmma_f32_16x16x32_bf16(
                            false, wfr[mt][ks].v, false, bfr[ks].v,
                            (short)0, acc, false, false);
                #pragma unroll
                for (int v = 0; v < 8; ++v) {
                    const int oc = mt * 16 + lhalf * 8 + v;
                    c2lds[oc * C2_LD + p] = f2bf(fmaxf(acc[v], 0.0f));
                }
            }
        }
        __builtin_amdgcn_wave_barrier();

        // ---- adaptive pool as GEMM: pooled = relu(D2) @ P ----
        #pragma unroll
        for (int mt = 0; mt < 4; ++mt) {
            v8f acc = {0.f, 0.f, 0.f, 0.f, 0.f, 0.f, 0.f, 0.f};
            #pragma unroll
            for (int ks = 0; ks < 3; ++ks) {
                BFrag a;
                const unsigned short* pr3 =
                    &c2lds[(mt * 16 + pcol) * C2_LD + ks * 32 + lhalf * 16];
                a.q[0] = *(const v4u*)pr3;
                a.q[1] = *(const v4u*)(pr3 + 8);
                acc = __builtin_amdgcn_wmma_f32_16x16x32_bf16(
                        false, a.v, false, pfr[ks].v, (short)0, acc, false, false);
            }
            if (pcol < 4) {
                #pragma unroll
                for (int v = 0; v < 8; ++v)
                    pooled[(mt * 16 + lhalf * 8 + v) * 4 + pcol] = acc[v] * 1.28f; // x32/25
            }
        }
        __builtin_amdgcn_wave_barrier();

        // ---- FC 256 -> 32 (lane = output feature) ----
        float a = feb[lane];
        for (int k = 0; k < 256; ++k) a += pooled[k] * few[k * 32 + lane];
        sigbf[(size_t)node * 32 + lane] = f2bf(a);
        __builtin_amdgcn_wave_barrier();
    }
}

// ---------------------------------------------------------------------------
// Dense GCN matmul: out[n x 64] = A_bf16[n x K] @ W_f32[K x 64], WMMA bf16.
// W fragments live in registers across all node tiles; grid-stride over tiles.
// ---------------------------------------------------------------------------
template <int K>
__global__ __launch_bounds__(128) void mm_kernel(
    const unsigned short* __restrict__ A, const float* __restrict__ W,
    float* __restrict__ out, int n)
{
    constexpr int KS = K / 32;
    const int lane  = threadIdx.x & 31;
    const int wave  = threadIdx.x >> 5;
    const int pcol  = lane & 15;
    const int lhalf = lane >> 4;

    BFrag bfr[4][KS];
    #pragma unroll
    for (int nt = 0; nt < 4; ++nt) {
        #pragma unroll
        for (int ks = 0; ks < KS; ++ks) {
            const int ncol = nt * 16 + pcol;
            const int kb   = ks * 32 + lhalf * 16;
            #pragma unroll
            for (int j = 0; j < 16; ++j)
                bfr[nt][ks].u[j] = f2bf(W[(kb + j) * 64 + ncol]);
        }
    }

    const int ntiles = (n + 15) >> 4;
    const int nwaves = gridDim.x * 4;
    for (int t = blockIdx.x * 4 + wave; t < ntiles; t += nwaves) {
        int row = t * 16 + pcol;
        if (row >= n) row = n - 1;
        const unsigned short* arow = A + (size_t)row * K;
        __builtin_prefetch(A + (size_t)(t + nwaves) * 16 * K, 0, 1);
        BFrag a[KS];
        #pragma unroll
        for (int ks = 0; ks < KS; ++ks) {
            const int kb = ks * 32 + lhalf * 16;
            a[ks].q[0] = *(const v4u*)(arow + kb);
            a[ks].q[1] = *(const v4u*)(arow + kb + 8);
        }
        #pragma unroll
        for (int nt = 0; nt < 4; ++nt) {
            v8f acc = {0.f, 0.f, 0.f, 0.f, 0.f, 0.f, 0.f, 0.f};
            #pragma unroll
            for (int ks = 0; ks < KS; ++ks)
                acc = __builtin_amdgcn_wmma_f32_16x16x32_bf16(
                        false, a[ks].v, false, bfr[nt][ks].v, (short)0, acc, false, false);
            #pragma unroll
            for (int v = 0; v < 8; ++v) {
                const int r = t * 16 + lhalf * 8 + v;
                if (r < n) out[(size_t)r * 64 + nt * 16 + pcol] = acc[v];
            }
        }
    }
}

// ---------------------------------------------------------------------------
// Degree / attention / GCN aggregation / head kernels (bandwidth-trivial)
// ---------------------------------------------------------------------------
__global__ void deg_init_kernel(float* deg, float* odeg, int n) {
    int i = blockIdx.x * blockDim.x + threadIdx.x;
    if (i < n) { deg[i] = 1.0f; odeg[i] = 0.0f; }   // self-loop pre-counted
}

__global__ void degree_kernel(const int* __restrict__ src, const int* __restrict__ dst,
                              float* deg, float* odeg, int e) {
    int i = blockIdx.x * blockDim.x + threadIdx.x;
    if (i < e) {
        atomicAdd(&deg[dst[i]], 1.0f);
        atomicAdd(&odeg[src[i]], 1.0f);
    }
}

__global__ void att_kernel(const float* __restrict__ aux, const int* __restrict__ src,
                           const int* __restrict__ dst, const float* __restrict__ odeg,
                           float* __restrict__ att1, float* __restrict__ att2, int e) {
    int i = blockIdx.x * blockDim.x + threadIdx.x;
    if (i >= e) return;
    const int s = src[i], d = dst[i];
    const float fd = fabsf(aux[s * 3 + 2] - aux[d * 3 + 2]);
    att1[i] = (fd == 1.0f) ? 1.0f / fmaxf(odeg[s], 1.0f) : 0.0f;
    const float as_ = aux[s * 3 + 0], rs = aux[s * 3 + 1];
    const float ad  = aux[d * 3 + 0], rd = aux[d * 3 + 1];
    const float dx = rd * cosf(ad) - rs * cosf(as_);
    const float dy = rd * sinf(ad) - rs * sinf(as_);
    const float fds = (fd == 2.0f) ? fd : 1.0f;
    const float vel = sqrtf(dx * dx + dy * dy + 1e-12f) / fds;
    att2[i] = (fd == 2.0f) ? expf(-vel * (1.0f / 8.5f)) : 0.0f;
}

__global__ void gcn_init_kernel(const float* __restrict__ h, const float* __restrict__ deg,
                                float* __restrict__ acc, int total) {
    int idx = blockIdx.x * blockDim.x + threadIdx.x;
    if (idx < total) acc[idx] = h[idx] / deg[idx >> 6];   // self-loop: dis_i^2 = 1/deg
}

__global__ void gcn_scatter_kernel(const int* __restrict__ src, const int* __restrict__ dst,
                                   const float* __restrict__ h, const float* __restrict__ deg,
                                   float* __restrict__ acc, int e) {
    int t = blockIdx.x * blockDim.x + threadIdx.x;
    int i = t >> 5;
    if (i >= e) return;
    const int f = (t & 31) * 2;
    const int s = src[i], d = dst[i];
    const float norm = rsqrtf(deg[s]) * rsqrtf(deg[d]);
    atomicAdd(&acc[(size_t)d * 64 + f],     h[(size_t)s * 64 + f]     * norm);
    atomicAdd(&acc[(size_t)d * 64 + f + 1], h[(size_t)s * 64 + f + 1] * norm);
}

__global__ void gcn_final_kernel(const float* __restrict__ acc, const float* __restrict__ b,
                                 const float* __restrict__ att, unsigned short* __restrict__ xbf,
                                 int total) {
    int idx = blockIdx.x * blockDim.x + threadIdx.x;
    if (idx < total) {
        const int i = idx >> 6, f = idx & 63;
        xbf[idx] = f2bf(fmaxf((acc[idx] + b[f]) * att[i], 0.0f));
    }
}

__global__ __launch_bounds__(128) void head_kernel(
    const unsigned short* __restrict__ x2, const float* __restrict__ fcw,
    const float* __restrict__ fcb, const float* __restrict__ ow,
    const float* __restrict__ ob, float* __restrict__ out, int n)
{
    const int lane = threadIdx.x & 31;
    const int node = blockIdx.x * 4 + (threadIdx.x >> 5);
    if (node >= n) return;
    float a = fcb[lane];
    for (int k = 0; k < 64; ++k)
        a += bf2f(x2[(size_t)node * 64 + k]) * fcw[k * 32 + lane];
    float c = fmaxf(a, 0.0f) * ow[lane];
    #pragma unroll
    for (int off = 16; off > 0; off >>= 1) c += __shfl_down(c, off, 32);
    if (lane == 0) out[node] = 1.0f / (1.0f + expf(-(c + ob[0])));
}

// ---------------------------------------------------------------------------
// Launcher
// ---------------------------------------------------------------------------
extern "C" void kernel_launch(void* const* d_in, const int* in_sizes, int n_in,
                              void* d_out, int out_size, void* d_ws, size_t ws_size,
                              hipStream_t stream)
{
    const float* x   = (const float*)d_in[0];
    const float* aux = (const float*)d_in[1];
    const int*   ei  = (const int*)d_in[2];
    const float* c1w = (const float*)d_in[3];
    const float* c1b = (const float*)d_in[4];
    const float* c2w = (const float*)d_in[5];
    const float* c2b = (const float*)d_in[6];
    const float* few = (const float*)d_in[7];
    const float* feb = (const float*)d_in[8];
    const float* g1w = (const float*)d_in[9];
    const float* g1b = (const float*)d_in[10];
    const float* g2w = (const float*)d_in[11];
    const float* g2b = (const float*)d_in[12];
    const float* fcw = (const float*)d_in[13];
    const float* fcb = (const float*)d_in[14];
    const float* ow  = (const float*)d_in[15];
    const float* ob  = (const float*)d_in[16];
    float* out = (float*)d_out;

    const int n = in_sizes[0] / 64;     // 50000 nodes
    const int e = in_sizes[2] / 2;      // 50000 edges
    const int* src = ei;
    const int* dst = ei + e;

    char* w = (char*)d_ws;
    auto take = [&](size_t bytes) {
        char* p = w;
        w += (bytes + 255) & ~(size_t)255;
        return p;
    };
    float*          deg   = (float*)take((size_t)n * 4);
    float*          odeg  = (float*)take((size_t)n * 4);
    float*          att1  = (float*)take((size_t)e * 4);
    float*          att2  = (float*)take((size_t)e * 4);
    unsigned short* sigbf = (unsigned short*)take((size_t)n * 32 * 2);
    float*          h     = (float*)take((size_t)n * 64 * 4);
    float*          acc   = (float*)take((size_t)n * 64 * 4);
    unsigned short* xbf   = (unsigned short*)take((size_t)n * 64 * 2);
    (void)ws_size; (void)out_size; (void)n_in;

    deg_init_kernel<<<(n + 255) / 256, 256, 0, stream>>>(deg, odeg, n);
    degree_kernel<<<(e + 255) / 256, 256, 0, stream>>>(src, dst, deg, odeg, e);

    int feblocks = 4096;
    if (feblocks > n) feblocks = n;
    fe_kernel<<<feblocks, 32, 0, stream>>>(x, c1w, c1b, c2w, c2b, few, feb, sigbf, n);
    att_kernel<<<(e + 255) / 256, 256, 0, stream>>>(aux, src, dst, odeg, att1, att2, e);

    const int ntiles   = (n + 15) / 16;
    const int mmblocks = (ntiles + 3) / 4;

    // GCN layer 1: h = sig @ W1 ; acc = A_hat h ; x1 = relu((acc+b)*att1)
    mm_kernel<32><<<mmblocks, 128, 0, stream>>>(sigbf, g1w, h, n);
    gcn_init_kernel<<<(n * 64 + 255) / 256, 256, 0, stream>>>(h, deg, acc, n * 64);
    gcn_scatter_kernel<<<(e * 32 + 255) / 256, 256, 0, stream>>>(src, dst, h, deg, acc, e);
    gcn_final_kernel<<<(n * 64 + 255) / 256, 256, 0, stream>>>(acc, g1b, att1, xbf, n * 64);

    // GCN layer 2
    mm_kernel<64><<<mmblocks, 128, 0, stream>>>(xbf, g2w, h, n);
    gcn_init_kernel<<<(n * 64 + 255) / 256, 256, 0, stream>>>(h, deg, acc, n * 64);
    gcn_scatter_kernel<<<(e * 32 + 255) / 256, 256, 0, stream>>>(src, dst, h, deg, acc, e);
    gcn_final_kernel<<<(n * 64 + 255) / 256, 256, 0, stream>>>(acc, g2b, att2, xbf, n * 64);

    // Head: relu(x2 @ fc + b) @ out_w -> sigmoid
    head_kernel<<<(n + 3) / 4, 128, 0, stream>>>(xbf, fcw, fcb, ow, ob, out, n);
}
